// coRNNCell2_27857157882574
// MI455X (gfx1250) — compile-verified, matrix operand order checked by reading, not run on hardware
//
#include <hip/hip_runtime.h>
#include <stdint.h>

// ---- problem constants ----
#define NBATCH 32
#define NHID   128
#define ICH    256      // 2*NHID
#define HH     12
#define WW     10
#define HW     120
#define IKL    2304     // ICH*9 reduction length
#define KSTEPS 72       // IKL / 32
#define NT     8        // o tiles of 16  (128/16)
#define MT     2        // b tiles of 16  (32/16)
#define FRAG   512      // bf16 elements per 16x16x32 fragment (32 lanes * 16)

typedef __attribute__((ext_vector_type(16))) __bf16 v16bf;
typedef __attribute__((ext_vector_type(8)))  float  v8f;
typedef __attribute__((ext_vector_type(4)))  unsigned int v4u;
typedef __attribute__((ext_vector_type(4)))  int v4i;

union FragU { v4u q[2]; v16bf v; };

__device__ __forceinline__ uint16_t f32_bf16(float f) {
  uint32_t u = __float_as_uint(f);
  u += 0x7fffu + ((u >> 16) & 1u);        // round-to-nearest-even
  return (uint16_t)(u >> 16);
}

// -------------------------------------------------------------------------
// Kernel 1: repack fp32 weights [o][ikl][uv] -> per-uv bf16 B-fragments.
// Block = (nt, ks, k8): 16 o x 8 kk x all 120 uv tile through LDS.
// Reads: 16 contiguous 960-float segments (full cachelines along uv).
// Writes: fragment-order bf16, dword-coalesced per uv.
// -------------------------------------------------------------------------
__global__ void __launch_bounds__(256) bpack_kernel(const float* __restrict__ w,
                                                    uint16_t* __restrict__ Bp) {
  const int nt = blockIdx.x & 7;
  const int ks = (blockIdx.x >> 3) % KSTEPS;
  const int k8 = blockIdx.x / (8 * KSTEPS);      // 0..3 (kk sub-octet)
  const int kkBase = ks * 32 + k8 * 8;
  const int oBase  = nt * 16;
  const int tid = threadIdx.x;

  __shared__ uint16_t tile[16 * 8 * 120];        // [o_l][kk_l][uv]  30.7 KB

  for (int it = 0; it < 60; ++it) {              // 15360 floats
    int flat = it * 256 + tid;
    int o_l  = flat / 960;
    int rem  = flat - o_l * 960;                 // kk_l*120 + uv (contiguous in mem)
    float v = w[(size_t)(oBase + o_l) * (IKL * HW) + (size_t)kkBase * HW + rem];
    tile[o_l * 960 + rem] = f32_bf16(v);
  }
  __syncthreads();

  const int o_l  = (tid >> 2) & 15;
  const int q    = tid & 3;                      // dword (2 kk) within octet
  const int uvo  = tid >> 6;                     // 0..3
  const int fbase = (((k8 & 1) * 16 + o_l) * 16 + (k8 >> 1) * 8);
  for (int j = 0; j < 120; j += 4) {
    int uv = j + uvo;
    uint32_t v0 = tile[(o_l * 8 + 2 * q) * 120 + uv];
    uint32_t v1 = tile[(o_l * 8 + 2 * q + 1) * 120 + uv];
    size_t fragBase = (size_t)((uv * NT + nt) * KSTEPS + ks) * FRAG;
    ((uint32_t*)(Bp + fragBase))[(fbase >> 1) + q] = v0 | (v1 << 16);
  }
}

// -------------------------------------------------------------------------
// Kernel 2: build bf16 A-fragments (3x3 zero-padded patches of concat(hz,hy)).
// One thread per output bf16 element; writes fully coalesced, reads hit L2.
// -------------------------------------------------------------------------
__global__ void __launch_bounds__(256) apack_kernel(const float* __restrict__ hy,
                                                    const float* __restrict__ hz,
                                                    uint16_t* __restrict__ Ap) {
  int g = blockIdx.x * 256 + threadIdx.x;        // 8,847,360 total
  int e    = g & 15;
  int lane = (g >> 4) & 31;
  int rest = g >> 9;
  int ks   = rest % KSTEPS;
  rest    /= KSTEPS;
  int mt = rest & 1;
  int uv = rest >> 1;

  int b = mt * 16 + (lane & 15);
  int kk = (lane < 16) ? (ks * 32 + (e < 8 ? e : e + 8))
                       : (ks * 32 + (e < 8 ? e + 8 : e + 16));
  int ich = kk / 9;
  int r9  = kk - ich * 9;
  int kr = r9 / 3, kl = r9 - kr * 3;
  int u = uv / 10, v = uv - u * 10;
  int uu = u + kr - 1, vv = v + kl - 1;

  float val = 0.f;
  if ((unsigned)uu < 12u && (unsigned)vv < 10u) {
    const float* src = (ich < 128) ? hz : hy;    // channels: [hz | hy]
    int c = ich & 127;
    val = src[((b * 128 + c) * 12 + uu) * 10 + vv];
  }
  Ap[g] = f32_bf16(val);
}

// -------------------------------------------------------------------------
// Kernel 3: per-pixel GEMM [32 x 2304] x [2304 x 128] via v_wmma_f32_16x16x32_bf16
// + fused coRNN update. 1 block / pixel, 8 waves; wave w owns o-tile w and both
// b-tiles. A chunks staged in LDS shared by all 8 waves.
//
// Async path: GLOBAL_LOAD_ASYNC_TO_LDS_B128 (ASYNCcnt-tracked, no VGPR round
// trip), double-buffered and overlapped with WMMA compute.
// -------------------------------------------------------------------------
#if defined(__has_builtin)
#if __has_builtin(__builtin_amdgcn_global_load_async_to_lds_b128)
#define USE_ASYNC_LDS 1
#endif
#endif

#ifdef USE_ASYNC_LDS

#define CS 12                                    // K-steps per chunk
#define NC (KSTEPS / CS)                         // 6 chunks
#define CHUNK_V4 (MT * CS * (FRAG / 8))          // 1536 v4 = 24 KB per buffer

typedef __attribute__((address_space(1))) v4i* gv4i_p;
typedef __attribute__((address_space(3))) v4i* lv4i_p;

__global__ void __launch_bounds__(256) gemm_kernel(const uint16_t* __restrict__ Ap,
                                                   const uint16_t* __restrict__ Bp,
                                                   const float* __restrict__ hy,
                                                   const float* __restrict__ hz,
                                                   const float* __restrict__ bias,
                                                   float* __restrict__ out) {
  const int uv   = blockIdx.x;
  const int tid  = threadIdx.x;
  const int wv   = tid >> 5;                     // wave id == o-tile (wave32)
  const int lane = tid & 31;

  __shared__ v4u smemA[2 * CHUNK_V4];            // 48 KB double buffer
  lv4i_p lds = (lv4i_p)smemA;

  v8f acc0 = {0.f,0.f,0.f,0.f,0.f,0.f,0.f,0.f};
  v8f acc1 = {0.f,0.f,0.f,0.f,0.f,0.f,0.f,0.f};

  const uint16_t* ApUV = Ap + (size_t)uv * (MT * KSTEPS * FRAG);
  const uint16_t* BpW  = Bp + (size_t)((uv * NT + wv) * KSTEPS) * FRAG;

  // async-copy chunk c of A fragments into LDS buffer `buf`
  auto issue = [&](int c, int buf) {
    #pragma unroll
    for (int it = 0; it < CHUNK_V4 / 256; ++it) {   // 6 x b128 per thread
      int flat = it * 256 + tid;
      int mt   = flat / (CHUNK_V4 / 2);
      int off  = flat - mt * (CHUNK_V4 / 2);
      gv4i_p src = (gv4i_p)(uint16_t*)(ApUV + (size_t)(mt * KSTEPS + c * CS) * FRAG);
      __builtin_amdgcn_global_load_async_to_lds_b128(
          src + off, lds + buf * CHUNK_V4 + mt * (CHUNK_V4 / 2) + off, 0, 0);
    }
  };

  issue(0, 0);
  for (int c = 0; c < NC; ++c) {
    asm volatile("s_wait_asynccnt 0" ::: "memory");  // my chunk-c (and older) done
    __syncthreads();                                 // everyone's chunk c visible,
                                                     // everyone done with buf (c+1)&1
    if (c + 1 < NC) issue(c + 1, (c + 1) & 1);       // overlap with compute below

    const int bufBase = (c & 1) * CHUNK_V4;
    for (int ksl = 0; ksl < CS; ++ksl) {
      FragU bf, a0, a1;
      const v4u* bsrc = (const v4u*)(BpW + (size_t)(c * CS + ksl) * FRAG) + lane * 2;
      bf.q[0] = bsrc[0]; bf.q[1] = bsrc[1];
      const v4u* as0 = smemA + bufBase + (0 * CS + ksl) * 64 + lane * 2;
      const v4u* as1 = smemA + bufBase + (1 * CS + ksl) * 64 + lane * 2;
      a0.q[0] = as0[0]; a0.q[1] = as0[1];
      a1.q[0] = as1[0]; a1.q[1] = as1[1];
      acc0 = __builtin_amdgcn_wmma_f32_16x16x32_bf16(false, a0.v, false, bf.v,
                                                     (short)0, acc0, false, false);
      acc1 = __builtin_amdgcn_wmma_f32_16x16x32_bf16(false, a1.v, false, bf.v,
                                                     (short)0, acc1, false, false);
    }
  }

#else  // ---------------- synchronous fallback (proven path) ----------------

#define CS 24

__global__ void __launch_bounds__(256) gemm_kernel(const uint16_t* __restrict__ Ap,
                                                   const uint16_t* __restrict__ Bp,
                                                   const float* __restrict__ hy,
                                                   const float* __restrict__ hz,
                                                   const float* __restrict__ bias,
                                                   float* __restrict__ out) {
  const int uv   = blockIdx.x;
  const int tid  = threadIdx.x;
  const int wv   = tid >> 5;
  const int lane = tid & 31;

  __shared__ v4u smemA[2 * CS * (FRAG / 8)];     // 3072 * 16B = 48 KB

  v8f acc0 = {0.f,0.f,0.f,0.f,0.f,0.f,0.f,0.f};
  v8f acc1 = {0.f,0.f,0.f,0.f,0.f,0.f,0.f,0.f};

  const uint16_t* ApUV = Ap + (size_t)uv * (MT * KSTEPS * FRAG);
  const uint16_t* BpW  = Bp + (size_t)((uv * NT + wv) * KSTEPS) * FRAG;

  for (int c = 0; c < KSTEPS / CS; ++c) {
    __syncthreads();
    for (int it = 0; it < 12; ++it) {            // 3072 x 16B cooperative copy
      int flat = it * 256 + tid;
      int mt   = flat / 1536;
      int off  = flat - mt * 1536;
      const v4u* src = (const v4u*)(ApUV + (size_t)(mt * KSTEPS + c * CS) * FRAG);
      smemA[mt * 1536 + off] = src[off];
    }
    __syncthreads();

    for (int ksl = 0; ksl < CS; ++ksl) {
      FragU bf, a0, a1;
      const v4u* bsrc = (const v4u*)(BpW + (size_t)(c * CS + ksl) * FRAG) + lane * 2;
      bf.q[0] = bsrc[0]; bf.q[1] = bsrc[1];
      const v4u* as0 = smemA + (0 * CS + ksl) * 64 + lane * 2;
      const v4u* as1 = smemA + (1 * CS + ksl) * 64 + lane * 2;
      a0.q[0] = as0[0]; a0.q[1] = as0[1];
      a1.q[0] = as1[0]; a1.q[1] = as1[1];
      acc0 = __builtin_amdgcn_wmma_f32_16x16x32_bf16(false, a0.v, false, bf.v,
                                                     (short)0, acc0, false, false);
      acc1 = __builtin_amdgcn_wmma_f32_16x16x32_bf16(false, a1.v, false, bf.v,
                                                     (short)0, acc1, false, false);
    }
  }

#endif // USE_ASYNC_LDS

  // Epilogue. C/D layout: lane<16 -> N=lane, M=r ; lane>=16 -> N=lane-16, M=r+8.
  const int o  = wv * 16 + (lane & 15);
  const int hi = (lane >> 4) << 3;               // 0 or 8
  const float bv = bias[o * HW + uv];
  float* hyOut = out;
  float* hzOut = out + (size_t)NBATCH * NHID * HW;

  for (int r = 0; r < 8; ++r) {
    #pragma unroll
    for (int half = 0; half < 2; ++half) {
      int b = half * 16 + r + hi;
      float conv = (half ? acc1[r] : acc0[r]) + bv;
      int idx = (b * 128 + o) * 120 + uv;
      float hyv = hy[idx];
      float hzv = hz[idx];
      float t   = tanhf(conv);
      float hzn = hzv + 0.01f * (t - 2.7f * hyv - 4.7f * hzv);
      float hyn = hyv + 0.01f * hzn;
      hyOut[idx] = hyn;
      hzOut[idx] = hzn;
    }
  }
}

// -------------------------------------------------------------------------
extern "C" void kernel_launch(void* const* d_in, const int* in_sizes, int n_in,
                              void* d_out, int out_size, void* d_ws, size_t ws_size,
                              hipStream_t stream) {
  const float* hy   = (const float*)d_in[0];
  const float* hz   = (const float*)d_in[1];
  const float* w    = (const float*)d_in[2];
  const float* bias = (const float*)d_in[3];

  const size_t nB = (size_t)HW * NT * KSTEPS * FRAG;   // 35,389,440 bf16
  const size_t nA = (size_t)HW * MT * KSTEPS * FRAG;   //  8,847,360 bf16
  if (ws_size < (nB + nA) * sizeof(uint16_t)) return;  // deterministic guard

  uint16_t* Bp = (uint16_t*)d_ws;
  uint16_t* Ap = Bp + nB;

  bpack_kernel<<<8 * KSTEPS * 4, 256, 0, stream>>>(w, Bp);
  apack_kernel<<<(int)(nA / 256), 256, 0, stream>>>(hy, hz, Ap);
  gemm_kernel<<<HW, 256, 0, stream>>>(Ap, Bp, hy, hz, bias, (float*)d_out);
}
